// LSTMActionAgent_103079215406
// MI455X (gfx1250) — matrix operand order repaired and checked by reading, not compiled, not run on hardware
//
#include <hip/hip_runtime.h>

// ---------------------------------------------------------------------------
// LSTM action agent for MI455X (gfx1250).
//   bf16 WMMA GEMM, LDS double-buffered via GLOBAL_LOAD_ASYNC_TO_LDS_B128
//   (ASYNCcnt), fused LSTM cell update: 2 dependent kernels per timestep.
//   B=512, S=256, I=12, H=256, A=9, G=4H=1024, K=512
// ---------------------------------------------------------------------------

typedef __attribute__((ext_vector_type(16))) __bf16 v16bf;
typedef __attribute__((ext_vector_type(8)))  __bf16 v8bf;
typedef __attribute__((ext_vector_type(8)))  float  v8f;

#define BB 512
#define SS 256
#define II 12
#define HH 256
#define AA 9
#define GG 1024          // 4*H
#define KK 512           // H (input part) + H (recurrent part)
#define BH (BB * HH)     // 131072

// ---------------- workspace layout (byte offsets, 256B aligned) ------------
#define OFF_EMB   ((size_t)0)                         // bf16 [B,S,H]   64 MB
#define OFF_WCAT  (OFF_EMB  + (size_t)BB*SS*HH*2)     // bf16 [2,G,K]    2 MB
#define OFF_BSUM  (OFF_WCAT + (size_t)2*GG*KK*2)      // f32  [2,G]      8 KB
#define OFF_HBF   (OFF_BSUM + (size_t)2*GG*4)         // bf16 [2,2,B,H]  1 MB (ping-pong)
#define OFF_HF32  (OFF_HBF  + (size_t)4*BH*2)         // f32  [2,B,H]    1 MB
#define OFF_CF32  (OFF_HF32 + (size_t)2*BH*4)         // f32  [2,B,H]    1 MB

// ---------------- weight packing (permuted rows: n' = h*4 + gate) ----------
__global__ void pack_weights(const float* __restrict__ w_ih0,
                             const float* __restrict__ w_hh0,
                             const float* __restrict__ w_ih1,
                             const float* __restrict__ w_hh1,
                             __bf16* __restrict__ Wcat) {
  int idx = blockIdx.x * blockDim.x + threadIdx.x;     // 2*1024*512
  int L   = idx >> 19;
  int rem = idx & ((1 << 19) - 1);
  int n2 = rem >> 9;                 // permuted row n'
  int k  = rem & 511;
  int g  = n2 & 3;                   // gate: 0=i,1=f,2=g,3=o
  int hh = n2 >> 2;                  // hidden unit
  int n  = g * HH + hh;              // original row in (4H, H) weights
  const float* wi = L ? w_ih1 : w_ih0;
  const float* wh = L ? w_hh1 : w_hh0;
  float v = (k < HH) ? wi[n * HH + k] : wh[n * HH + (k - HH)];
  Wcat[idx] = (__bf16)v;
}

__global__ void pack_bias(const float* __restrict__ bi0, const float* __restrict__ bh0,
                          const float* __restrict__ bi1, const float* __restrict__ bh1,
                          float* __restrict__ bsum) {
  int idx = blockIdx.x * blockDim.x + threadIdx.x;     // 2048
  int L = idx >> 10, n2 = idx & 1023;
  int g = n2 & 3, hh = n2 >> 2;
  int n = g * HH + hh;
  bsum[idx] = L ? (bi1[n] + bh1[n]) : (bi0[n] + bh0[n]);
}

// ---------------- embedding: emb = relu(x @ W_emb^T + b), bf16 -------------
__global__ void emb_kernel(const float* __restrict__ x,
                           const float* __restrict__ W,
                           const float* __restrict__ bias,
                           __bf16* __restrict__ emb) {
  int bs = blockIdx.x;          // one (b,s) per block
  int h  = threadIdx.x;         // 256 threads
  const float* xr = x + (size_t)bs * II;
  const float* wr = W + (size_t)h * II;
  float s = bias[h];
#pragma unroll
  for (int i = 0; i < II; ++i) s += xr[i] * wr[i];
  s = fmaxf(s, 0.0f);
  emb[(size_t)bs * HH + h] = (__bf16)s;
}

// ---------------- fused timestep kernel ------------------------------------
// Block tile: M=64 (batch) x N=128 (4 gates x 32 hidden), K=512, BK=32,
// double-buffered in LDS via async global->LDS DMA. 8 waves, each a 32x32
// tile (2x2 WMMA accums). Epilogue: gate tile -> LDS (union overlay) ->
// in-block LSTM cell update.
__device__ __forceinline__ float sigf(float x) { return 1.0f / (1.0f + __expf(-x)); }

struct StageSM {
  __bf16 As[2][64][40];    // BK=32, rows padded to 40 (80B, 16B-multiple)
  __bf16 Bs[2][128][40];
};
union alignas(16) FusedSM {
  StageSM s;
  float Gt[64][132];       // 64 x 128 gate tile, padded row
};

// LDS byte offset of a shared-memory pointer (AS3 pointers are 32-bit).
__device__ __forceinline__ uint32_t lds_off(const void* p) {
  return (uint32_t)(uintptr_t)(__attribute__((address_space(3))) const void*)p;
}

// Async 16B global -> LDS copy, tracked by ASYNCcnt.
__device__ __forceinline__ void async_copy_b128(uint32_t lds, const __bf16* g) {
  asm volatile("global_load_async_to_lds_b128 %0, %1, off"
               :: "v"(lds), "v"(g)
               : "memory");
}

__device__ __forceinline__ void wait_asynccnt0() {
  asm volatile("s_wait_asynccnt 0x0" ::: "memory");
}

__global__ __launch_bounds__(256)
void lstm_step_fused(const __bf16* __restrict__ Alow, int astride,
                     const __bf16* __restrict__ Ahigh,      // h_prev bf16 (read)
                     const __bf16* __restrict__ Wc,          // packed [G,K]
                     const float* __restrict__ bsumP,        // packed [G]
                     float* __restrict__ c_st,
                     float* __restrict__ h_st,
                     __bf16* __restrict__ h_bf_out) {        // h_new bf16 (write)
  __shared__ FusedSM sm;

  const int tid  = threadIdx.x;
  const int lane = tid & 31;
  const int r    = lane & 15;
  const int half = lane >> 4;
  const int wv   = tid >> 5;           // 0..7
  const int wm   = wv >> 2;            // 0..1  (32-row M subtile)
  const int wn   = wv & 3;             // 0..3  (32-col N subtile)

  const int bm0 = (blockIdx.x >> 3) * 64;    // 8 M blocks
  const int bn0 = (blockIdx.x & 7) * 128;    // 8 N blocks

  // staging assignments: A slab 64x32 (8 bf16/thread), B slab 128x32 (16/thread)
  const int arow = tid >> 2;           // 0..63
  const int acol = (tid & 3) * 8;      // 0,8,16,24
  const int brow = tid >> 1;           // 0..127
  const int bcol = (tid & 1) * 16;     // 0,16

  // Issue async DMA of one K-slab into LDS buffer `buf`.
  auto stage = [&](int buf, int klo) {
    const int k = klo + acol;          // 8-chunk never straddles H boundary
    const __bf16* ap = (k < HH) ? (Alow + (size_t)(bm0 + arow) * (size_t)astride + k)
                                : (Ahigh + (size_t)(bm0 + arow) * HH + (k - HH));
    async_copy_b128(lds_off(&sm.s.As[buf][arow][acol]), ap);
    const __bf16* bp = Wc + (size_t)(bn0 + brow) * KK + klo + bcol;
    async_copy_b128(lds_off(&sm.s.Bs[buf][brow][bcol]), bp);
    async_copy_b128(lds_off(&sm.s.Bs[buf][brow][bcol + 8]), bp + 8);
  };

  // prologue: stage 0
  stage(0, 0);
  wait_asynccnt0();
  __syncthreads();

  v8f acc[2][2] = {};

  for (int kb = 0; kb < 16; ++kb) {
    const int cur = kb & 1;
    if (kb < 15)                       // async-fill next buffer while we compute
      stage(cur ^ 1, (kb + 1) * 32);

    // fragments from LDS (ISA bf16 A/B lane layouts)
    union { v16bf v; v8bf h[2]; } afr[2], bfr[2];
#pragma unroll
    for (int i = 0; i < 2; ++i) {
      const __bf16* base = &sm.s.As[cur][wm * 32 + i * 16 + r][0];
      afr[i].h[0] = *(const v8bf*)(base + half * 8);
      afr[i].h[1] = *(const v8bf*)(base + half * 8 + 16);
    }
#pragma unroll
    for (int j = 0; j < 2; ++j) {
      const __bf16* base = &sm.s.Bs[cur][wn * 32 + j * 16 + r][0];
      bfr[j].h[0] = *(const v8bf*)(base + half * 8);
      bfr[j].h[1] = *(const v8bf*)(base + half * 8 + 16);
    }
#pragma unroll
    for (int i = 0; i < 2; ++i)
#pragma unroll
      for (int j = 0; j < 2; ++j)
        acc[i][j] = __builtin_amdgcn_wmma_f32_16x16x32_bf16(
            false, afr[i].v, false, bfr[j].v, (short)0, acc[i][j], false, false);

    if (kb < 15) {                     // publish next slab to the workgroup
      wait_asynccnt0();
      __syncthreads();
    }
  }

  // ---- epilogue: gate tile to LDS (overlaid on staging buffers) ----
  __syncthreads();                     // all waves done reading stage
#pragma unroll
  for (int j = 0; j < 2; ++j) {
    const int nl = wn * 32 + j * 16 + r;
    const float bs = bsumP[bn0 + nl];
#pragma unroll
    for (int i = 0; i < 2; ++i) {
#pragma unroll
      for (int e = 0; e < 8; ++e) {
        const int ml = wm * 32 + i * 16 + half * 8 + e;
        sm.Gt[ml][nl] = acc[i][j][e] + bs;
      }
    }
  }
  __syncthreads();

  // ---- fused LSTM cell update over this block's (64 batch x 32 hidden) ----
  const int hbase = bn0 >> 2;          // h range start
#pragma unroll
  for (int p = 0; p < 8; ++p) {
    const int q  = tid + p * 256;      // 0..2047
    const int ml = q >> 5;             // 0..63 local batch row
    const int hl = q & 31;             // 0..31 local hidden unit
    const float ig = sm.Gt[ml][hl * 4 + 0];
    const float fg = sm.Gt[ml][hl * 4 + 1];
    const float gg = sm.Gt[ml][hl * 4 + 2];
    const float og = sm.Gt[ml][hl * 4 + 3];
    const size_t cidx = (size_t)(bm0 + ml) * HH + (hbase + hl);
    float c = c_st[cidx];
    c = sigf(fg) * c + sigf(ig) * tanhf(gg);
    const float h = sigf(og) * tanhf(c);
    c_st[cidx] = c;
    h_st[cidx] = h;
    h_bf_out[cidx] = (__bf16)h;
  }
}

// ---------------- head: layernorm -> 256x32 relu -> 32x9 + mask ------------
__global__ void head_kernel(const float* __restrict__ h1,
                            const float* __restrict__ ln_g, const float* __restrict__ ln_b,
                            const float* __restrict__ w1, const float* __restrict__ b1,
                            const float* __restrict__ w2, const float* __restrict__ b2,
                            const float* __restrict__ mask,
                            float* __restrict__ out_logits) {
  __shared__ float sd[256];
  __shared__ float snorm[256];
  __shared__ float shid[32];
  __shared__ float smu, srstd;
  const int b = blockIdx.x, tid = threadIdx.x;
  float v = h1[(size_t)b * HH + tid];
  sd[tid] = v;
  __syncthreads();
  for (int s = 128; s > 0; s >>= 1) { if (tid < s) sd[tid] += sd[tid + s]; __syncthreads(); }
  if (tid == 0) smu = sd[0] * (1.0f / HH);
  __syncthreads();
  const float mu = smu;
  const float d = v - mu;
  sd[tid] = d * d;
  __syncthreads();
  for (int s = 128; s > 0; s >>= 1) { if (tid < s) sd[tid] += sd[tid + s]; __syncthreads(); }
  if (tid == 0) srstd = rsqrtf(sd[0] * (1.0f / HH) + 1e-5f);
  __syncthreads();
  snorm[tid] = d * srstd * ln_g[tid] + ln_b[tid];
  __syncthreads();
  if (tid < 32) {
    float s = b1[tid];
    const float* wr = w1 + (size_t)tid * HH;
    for (int k = 0; k < HH; ++k) s += snorm[k] * wr[k];
    shid[tid] = fmaxf(s, 0.0f);
  }
  __syncthreads();
  if (tid < AA) {
    float s = b2[tid];
    const float* wr = w2 + (size_t)tid * 32;
    for (int k = 0; k < 32; ++k) s += shid[k] * wr[k];
    s += (1.0f - mask[(size_t)b * AA + tid]) * -1e9f;
    out_logits[(size_t)b * AA + tid] = s;
  }
}

// ---------------- pack hidden state outputs: [h0|h1|c0|c1] -----------------
__global__ void copy_hidden(const float* __restrict__ h0, const float* __restrict__ h1,
                            const float* __restrict__ c0, const float* __restrict__ c1,
                            float* __restrict__ out) {
  int idx = blockIdx.x * blockDim.x + threadIdx.x;   // 4*B*H
  int which = idx >> 17;
  int off   = idx & (BH - 1);
  const float* src = (which == 0) ? h0 : (which == 1) ? h1 : (which == 2) ? c0 : c1;
  out[BB * AA + idx] = src[off];
}

// ---------------------------------------------------------------------------
extern "C" void kernel_launch(void* const* d_in, const int* in_sizes, int n_in,
                              void* d_out, int out_size, void* d_ws, size_t ws_size,
                              hipStream_t stream) {
  const float* x     = (const float*)d_in[0];
  const float* mask  = (const float*)d_in[1];
  const float* W_emb = (const float*)d_in[2];
  const float* b_emb = (const float*)d_in[3];
  const float* w_ih0 = (const float*)d_in[4];
  const float* w_hh0 = (const float*)d_in[5];
  const float* b_ih0 = (const float*)d_in[6];
  const float* b_hh0 = (const float*)d_in[7];
  const float* w_ih1 = (const float*)d_in[8];
  const float* w_hh1 = (const float*)d_in[9];
  const float* b_ih1 = (const float*)d_in[10];
  const float* b_hh1 = (const float*)d_in[11];
  const float* ln_g  = (const float*)d_in[12];
  const float* ln_b  = (const float*)d_in[13];
  const float* w1    = (const float*)d_in[14];
  const float* b1    = (const float*)d_in[15];
  const float* w2    = (const float*)d_in[16];
  const float* b2    = (const float*)d_in[17];
  float* out = (float*)d_out;

  char* ws = (char*)d_ws;
  __bf16* emb   = (__bf16*)(ws + OFF_EMB);
  __bf16* Wcat  = (__bf16*)(ws + OFF_WCAT);
  float*  bsum  = (float*)(ws + OFF_BSUM);
  __bf16* h_bf  = (__bf16*)(ws + OFF_HBF);   // [layer][pingpong][B*H]
  float*  h_f32 = (float*)(ws + OFF_HF32);
  float*  c_f32 = (float*)(ws + OFF_CF32);

  __bf16* hbf0[2] = { h_bf,            h_bf + BH };
  __bf16* hbf1[2] = { h_bf + 2 * BH,   h_bf + 3 * BH };
  float*  h0f = h_f32;
  float*  h1f = h_f32 + BH;
  float*  c0f = c_f32;
  float*  c1f = c_f32 + BH;
  __bf16* Wcat0 = Wcat;
  __bf16* Wcat1 = Wcat + (size_t)GG * KK;
  float*  bsum0 = bsum;
  float*  bsum1 = bsum + GG;

  // zero initial states (workspace is poisoned by the harness)
  hipMemsetAsync(h_bf, 0, (size_t)4 * BH * sizeof(__bf16), stream);
  hipMemsetAsync(c_f32, 0, (size_t)2 * BH * sizeof(float), stream);

  pack_weights<<<(2 * GG * KK) / 256, 256, 0, stream>>>(w_ih0, w_hh0, w_ih1, w_hh1, Wcat);
  pack_bias<<<8, 256, 0, stream>>>(b_ih0, b_hh0, b_ih1, b_hh1, bsum);
  emb_kernel<<<BB * SS, 256, 0, stream>>>(x, W_emb, b_emb, emb);

  for (int t = 0; t < SS; ++t) {
    const int pp = t & 1;        // read (prev h)
    const int qq = pp ^ 1;       // write (new h)
    // layer 0: A = [emb_t | h0_prev]
    lstm_step_fused<<<64, 256, 0, stream>>>(emb + (size_t)t * HH, SS * HH,
                                            hbf0[pp], Wcat0, bsum0,
                                            c0f, h0f, hbf0[qq]);
    // layer 1: A = [h0_t | h1_prev]
    lstm_step_fused<<<64, 256, 0, stream>>>(hbf0[qq], HH,
                                            hbf1[pp], Wcat1, bsum1,
                                            c1f, h1f, hbf1[qq]);
  }

  head_kernel<<<BB, 256, 0, stream>>>(h1f, ln_g, ln_b, w1, b1, w2, b2, mask, out);
  copy_hidden<<<(4 * BH) / 256, 256, 0, stream>>>(h0f, h1f, c0f, c1f, out);
}
